// SimilarityTreeLSTM_29154238005597
// MI455X (gfx1250) — matrix operand order repaired and checked by reading, not compiled, not run on hardware
//
#include <hip/hip_runtime.h>
#include <hip/hip_bf16.h>
#include <math.h>
#include <stdint.h>

// ---------------------------------------------------------------------------
// Types for CDNA5 WMMA (gfx1250, wave32)
// ---------------------------------------------------------------------------
typedef __attribute__((ext_vector_type(16))) __bf16 v16bf;
typedef __attribute__((ext_vector_type(8)))  __bf16 v8bf;
typedef __attribute__((ext_vector_type(8)))  float  v8f;

#define HID   1024
#define GATES 4096   // [i|o|u|f] each 1024
#define SEQ   512
#define NT_N  256    // GATES/16 N-tiles
#define NWG   128    // persistent workgroups in the scan kernel

// ===========================================================================
// Prep kernels: bf16 conversion / packing / gather / state init
// ===========================================================================

// Pack [Wioux|Wfx] (1024 x 4096, f32) into WMMA B-fragment order, bf16.
// Dest index = (((kt*NT_N)+nt)*32 + lane)*16 + e
//   half = lane>>4 ; k = kt*32 + half*8 + (e&7) + (e>>3)*16 ; n = nt*16 + (lane&15)
__global__ void pack_wx_kernel(const float* __restrict__ Wioux,
                               const float* __restrict__ Wfx,
                               __bf16* __restrict__ WxP) {
  size_t d = (size_t)blockIdx.x * blockDim.x + threadIdx.x;
  if (d >= (size_t)HID * GATES) return;
  int e    = (int)(d & 15);
  size_t r = d >> 4;
  int lane = (int)(r & 31); r >>= 5;
  int nt   = (int)(r & (NT_N - 1));
  int kt   = (int)(r >> 8);
  int half = lane >> 4;
  int k = kt * 32 + half * 8 + (e & 7) + ((e >> 3) << 4);
  int n = nt * 16 + (lane & 15);
  float v = (n < 3072) ? Wioux[(size_t)k * 3072 + n]
                       : Wfx[(size_t)k * 1024 + (n - 3072)];
  WxP[d] = (__bf16)v;
}

// Recurrent weights [Wiouh|Wfh] transposed to [col][k] bf16 (contiguous k per
// output column -> vector loads in the scan).  Also build the fused X bias.
__global__ void conv_wh_kernel(const float* __restrict__ Wiouh,
                               const float* __restrict__ Wfh,
                               const float* __restrict__ bioux,
                               const float* __restrict__ biouh,
                               const float* __restrict__ bfx,
                               const float* __restrict__ bfh,
                               __bf16* __restrict__ WhT,
                               float* __restrict__ biasX) {
  size_t d = (size_t)blockIdx.x * blockDim.x + threadIdx.x;
  if (d >= (size_t)HID * GATES) return;
  int k = (int)(d & (HID - 1));
  int n = (int)(d >> 10);
  float v = (n < 3072) ? Wiouh[(size_t)k * 3072 + n]
                       : Wfh[(size_t)k * 1024 + (n - 3072)];
  WhT[d] = (__bf16)v;
  if (d < GATES) {
    int c = (int)d;
    biasX[c] = (c < 3072) ? (bioux[c] + biouh[c]) : (bfx[c - 3072] + bfh[c - 3072]);
  }
}

// Gather token embeddings to bf16 activations; zero scan state & barrier.
__global__ void gather_init_kernel(const float* __restrict__ emb,
                                   const int* __restrict__ lin,
                                   const int* __restrict__ rin,
                                   __bf16* __restrict__ X2,  // [2][512][1024]
                                   float* __restrict__ hbuf, // [2][4][1024]
                                   unsigned* __restrict__ ctr) {
  size_t d = (size_t)blockIdx.x * blockDim.x + threadIdx.x;
  if (d < 2 * 4 * HID) hbuf[d] = 0.0f;
  if (d == 0) *ctr = 0u;
  if (d >= (size_t)2 * SEQ * HID) return;
  int k = (int)(d & (HID - 1));
  int t = (int)((d >> 10) & (SEQ - 1));
  int side = (int)(d >> 19);
  int tok = side ? rin[t] : lin[t];
  X2[d] = (__bf16)emb[(size_t)tok * HID + k];
}

// ===========================================================================
// WMMA GEMM: PX[side] = X[side] @ [Wioux|Wfx] + biasX      (bf16 -> f32)
// One 16x16 C tile per wave; K=1024 in 32-wide slices.
// ===========================================================================
__global__ void xproj_gemm_kernel(const __bf16* __restrict__ X2,  // [2][512][1024]
                                  const __bf16* __restrict__ WxP, // packed B frags
                                  const float* __restrict__ biasX,
                                  float* __restrict__ PX2) {      // [2][512][4096]
  const int lane = threadIdx.x & 31;
  const int wave = threadIdx.x >> 5;
  int tile = blockIdx.x * 8 + wave;          // 2 * 32 * 256 = 16384 tiles
  const int side = tile >> 13; tile &= 8191;
  const int mt = tile >> 8;                  // 0..31
  const int nt = tile & (NT_N - 1);          // 0..255
  const int half = lane >> 4;
  const int m = (mt << 4) + (lane & 15);

  const __bf16* arow = X2 + (size_t)side * SEQ * HID + (size_t)m * HID + half * 8;
  const __bf16* bptr = WxP + (((size_t)nt * 32 + lane) << 4);

  v8f acc = {};
  #pragma unroll 4
  for (int kt = 0; kt < 32; ++kt) {
    v8bf alo = *(const v8bf*)(arow + kt * 32);        // K = kt*32 + half*8 + 0..7
    v8bf ahi = *(const v8bf*)(arow + kt * 32 + 16);   // K = kt*32+16 + half*8 + 0..7
    v16bf a = __builtin_shufflevector(alo, ahi, 0, 1, 2, 3, 4, 5, 6, 7,
                                      8, 9, 10, 11, 12, 13, 14, 15);
    v16bf b = *(const v16bf*)(bptr + (size_t)kt * (NT_N * 32 * 16));
    acc = __builtin_amdgcn_wmma_f32_16x16x32_bf16(
        /*neg_a=*/false, a, /*neg_b=*/false, b,
        /*c_mod=*/(short)0, acc, /*reuse_a=*/false, /*reuse_b=*/false);
  }

  const int n = (nt << 4) + (lane & 15);
  const float bv = biasX[n];
  float* out = PX2 + (size_t)side * SEQ * GATES + n;
  #pragma unroll
  for (int v = 0; v < 8; ++v) {
    int mrow = (mt << 4) + v + (half << 3);   // C layout: VGPR v -> M = half*8 + v
    out[(size_t)mrow * GATES] = acc[v] + bv;
  }
}

// ===========================================================================
// Persistent recurrent scan: 4 batched chains (Lfwd, Lbwd, Rfwd, Rbwd) share
// one bf16 weight stream per step (L2-bandwidth-bound critical path).
// 128 WGs x 256 threads; WG owns 8 hidden units; global phase barrier / step.
// h-exchange uses CDNA5 async global->LDS copies (ASYNCcnt).
// ===========================================================================
__global__ void lstm_scan_kernel(const __bf16* __restrict__ WhT, // [4096][1024]
                                 const float* __restrict__ PXL,  // [512][4096]
                                 const float* __restrict__ PXR,
                                 float* __restrict__ hbuf,       // [2][4][1024]
                                 float* __restrict__ hFL, float* __restrict__ hBL,
                                 float* __restrict__ hFR, float* __restrict__ hBR,
                                 unsigned* __restrict__ ctr) {
  __shared__ float sh_h[4][HID];     // 16 KB   previous h, all 4 chains
  __shared__ float red[8][32][4];    // 4 KB    k-group partials
  __shared__ float sh_a[4][32];      // gate pre-activations for this WG
  __shared__ float c_state[4][8];    // persistent cell state

  const int tid = threadIdx.x;
  const int jbase = blockIdx.x * 8;        // hidden units owned by this WG
  const int col_local = tid & 31;          // 4 gates x 8 units
  const int gate = col_local >> 3;
  const int jj = col_local & 7;
  const int col = gate * HID + jbase + jj; // column in [0,4096)
  const int kg = tid >> 5;                 // k-group 0..7 (128 k each)

  if (tid < 32) c_state[tid >> 3][tid & 7] = 0.0f;

  const __bf16* wrow = WhT + ((size_t)col << 10) + (kg << 7);
  // Warm L2 with this thread's slice of the recurrent weight stream
  // (re-read every one of the 512 steps) -> global_prefetch_b8.
  __builtin_prefetch(wrow, 0, 3);

  // LDS byte offset of sh_h (generic LDS pointers carry the offset in the
  // low 32 bits of the address).
  const unsigned lds_h_base = (unsigned)(uintptr_t)(&sh_h[0][0]);

  for (int t = 0; t < SEQ; ++t) {
    const float* hsrc = hbuf + (size_t)(t & 1) * 4 * HID;
    // Async copy 16 KB of previous-step h (all 4 chains) global -> LDS.
    // 256 threads x 4 x B128 = 4096 floats; tracked by ASYNCcnt.
    #pragma unroll
    for (int q = 0; q < 4; ++q) {
      const int idx = tid + q * 256;                       // float4 index
      const unsigned ldst = lds_h_base + (unsigned)idx * 16u;
      const uint64_t gsrc = (uint64_t)(uintptr_t)((const char*)hsrc + (size_t)idx * 16);
      asm volatile("global_load_async_to_lds_b128 %0, %1, off"
                   :: "v"(ldst), "v"(gsrc)
                   : "memory");
    }
    __builtin_amdgcn_s_wait_asynccnt(0);
    __syncthreads();

    float a0 = 0.f, a1 = 0.f, a2 = 0.f, a3 = 0.f;
    #pragma unroll
    for (int kk = 0; kk < 128; kk += 8) {
      v8bf wv = *(const v8bf*)(wrow + kk);
      #pragma unroll
      for (int i = 0; i < 8; ++i) {
        float w = (float)wv[i];
        int k = (kg << 7) + kk + i;
        a0 += w * sh_h[0][k];
        a1 += w * sh_h[1][k];
        a2 += w * sh_h[2][k];
        a3 += w * sh_h[3][k];
      }
    }
    red[kg][col_local][0] = a0; red[kg][col_local][1] = a1;
    red[kg][col_local][2] = a2; red[kg][col_local][3] = a3;
    __syncthreads();

    if (tid < 32) {
      float s0 = 0.f, s1 = 0.f, s2 = 0.f, s3 = 0.f;
      #pragma unroll
      for (int g = 0; g < 8; ++g) {
        s0 += red[g][tid][0]; s1 += red[g][tid][1];
        s2 += red[g][tid][2]; s3 += red[g][tid][3];
      }
      const int cc = (tid >> 3) * HID + jbase + (tid & 7);
      s0 += PXL[(size_t)t * GATES + cc];
      s1 += PXL[(size_t)(SEQ - 1 - t) * GATES + cc];
      s2 += PXR[(size_t)t * GATES + cc];
      s3 += PXR[(size_t)(SEQ - 1 - t) * GATES + cc];
      sh_a[0][tid] = s0; sh_a[1][tid] = s1; sh_a[2][tid] = s2; sh_a[3][tid] = s3;
    }
    __syncthreads();

    if (tid < 8) {
      float* hdst = hbuf + (size_t)((t + 1) & 1) * 4 * HID;
      const int j = jbase + tid;
      #pragma unroll
      for (int ch = 0; ch < 4; ++ch) {
        float iv = sh_a[ch][tid],      ov = sh_a[ch][8 + tid];
        float uv = sh_a[ch][16 + tid], fv = sh_a[ch][24 + tid];
        float ig = 1.f / (1.f + __expf(-iv));
        float og = 1.f / (1.f + __expf(-ov));
        float ug = tanhf(uv);
        float fg = 1.f / (1.f + __expf(-fv));
        float c = ig * ug + fg * c_state[ch][tid];
        c_state[ch][tid] = c;
        float h = og * tanhf(c);
        hdst[ch * HID + j] = h;
        float* hout = (ch == 0) ? hFL : (ch == 1) ? hBL : (ch == 2) ? hFR : hBR;
        hout[(size_t)t * HID + j] = h;
      }
    }
    __syncthreads();

    if (tid == 0) {                        // device-wide phase barrier
      __threadfence();
      atomicAdd(ctr, 1u);
      const unsigned target = (unsigned)(t + 1) * NWG;
      while (atomicAdd(ctr, 0u) < target) __builtin_amdgcn_s_sleep(1);
    }
    __syncthreads();
  }
}

// ===========================================================================
// h = tanh(hf + reverse(hb))
// ===========================================================================
__global__ void combine_kernel(const float* __restrict__ hF,
                               const float* __restrict__ hB,
                               float* __restrict__ hOut) {
  size_t idx = (size_t)blockIdx.x * blockDim.x + threadIdx.x;
  if (idx >= (size_t)SEQ * HID) return;
  int t = (int)(idx >> 10);
  int j = (int)(idx & (HID - 1));
  hOut[idx] = tanhf(hF[idx] + hB[(size_t)(SEQ - 1 - t) * HID + j]);
}

// ===========================================================================
// Tail: attention rows that matter (only last rows of vl/vr are consumed),
// classifier, log-softmax.  One 1024-thread workgroup.
// ===========================================================================
__global__ void tail_kernel(const float* __restrict__ hl,
                            const float* __restrict__ hr,
                            const float* __restrict__ Wattn,
                            const float* __restrict__ battn,
                            const float* __restrict__ Wh,
                            const float* __restrict__ bwh,
                            const float* __restrict__ Wp,
                            const float* __restrict__ bwp,
                            float* __restrict__ out) {
  __shared__ float srow[SEQ], scol[SEQ];
  __shared__ float beta[HID], alpha[HID];
  __shared__ float lh[HID], rh[HID];
  __shared__ float vec[2 * HID];
  __shared__ float hid[512];
  __shared__ float inv[2];
  const int tid = threadIdx.x;
  const float* hl_last = hl + (size_t)(SEQ - 1) * HID;
  const float* hr_last = hr + (size_t)(SEQ - 1) * HID;

  if (tid < SEQ) {      // S row 511 and S column 511
    float s0 = 0.f, s1 = 0.f;
    const float* hri = hr + (size_t)tid * HID;
    const float* hli = hl + (size_t)tid * HID;
    for (int k = 0; k < HID; ++k) { s0 += hl_last[k] * hri[k]; s1 += hli[k] * hr_last[k]; }
    srow[tid] = s0; scol[tid] = s1;
  }
  __syncthreads();
  if (tid == 0) {       // softmax (serial over 512; negligible)
    float m0 = -1e30f, m1 = -1e30f;
    for (int i = 0; i < SEQ; ++i) { m0 = fmaxf(m0, srow[i]); m1 = fmaxf(m1, scol[i]); }
    float z0 = 0.f, z1 = 0.f;
    for (int i = 0; i < SEQ; ++i) {
      srow[i] = __expf(srow[i] - m0); z0 += srow[i];
      scol[i] = __expf(scol[i] - m1); z1 += scol[i];
    }
    inv[0] = 1.f / z0; inv[1] = 1.f / z1;
  }
  __syncthreads();
  {                     // beta[-1], alpha[-1]
    float b = 0.f, al = 0.f;
    for (int i = 0; i < SEQ; ++i) {
      b  += srow[i] * hr[(size_t)i * HID + tid];
      al += scol[i] * hl[(size_t)i * HID + tid];
    }
    beta[tid] = b * inv[0]; alpha[tid] = al * inv[1];
  }
  __syncthreads();
  {                     // lh, rh = [h_last, attn] @ Wattn + battn
    float a = 0.f, b = 0.f;
    for (int k = 0; k < HID; ++k) {
      float w0 = Wattn[(size_t)k * HID + tid];
      float w1 = Wattn[(size_t)(HID + k) * HID + tid];
      a += hl_last[k] * w0 + beta[k]  * w1;
      b += hr_last[k] * w0 + alpha[k] * w1;
    }
    lh[tid] = a + battn[tid]; rh[tid] = b + battn[tid];
  }
  __syncthreads();
  vec[tid]       = lh[tid] * rh[tid];
  vec[HID + tid] = fabsf(lh[tid] - rh[tid]);
  __syncthreads();
  if (tid < 512) {
    float a = bwh[tid];
    for (int n = 0; n < 2 * HID; ++n) a += vec[n] * Wh[(size_t)n * 512 + tid];
    hid[tid] = 1.f / (1.f + __expf(-a));
  }
  __syncthreads();
  if (tid == 0) {
    float logit[5];
    for (int c = 0; c < 5; ++c) {
      float a = bwp[c];
      for (int m = 0; m < 512; ++m) a += hid[m] * Wp[m * 5 + c];
      logit[c] = a;
    }
    float mx = logit[0];
    for (int c = 1; c < 5; ++c) mx = fmaxf(mx, logit[c]);
    float z = 0.f;
    for (int c = 0; c < 5; ++c) z += __expf(logit[c] - mx);
    float lz = mx + __logf(z);
    for (int c = 0; c < 5; ++c) out[c] = logit[c] - lz;
  }
}

// ===========================================================================
// Host launcher
// ===========================================================================
static inline char* carve(char*& p, size_t bytes) {
  char* r = p;
  p += (bytes + 255) & ~(size_t)255;
  return r;
}

extern "C" void kernel_launch(void* const* d_in, const int* in_sizes, int n_in,
                              void* d_out, int out_size, void* d_ws, size_t ws_size,
                              hipStream_t stream) {
  const float* emb   = (const float*)d_in[0];
  const float* Wioux = (const float*)d_in[1];
  const float* bioux = (const float*)d_in[2];
  const float* Wiouh = (const float*)d_in[3];
  const float* biouh = (const float*)d_in[4];
  const float* Wfx   = (const float*)d_in[5];
  const float* bfx   = (const float*)d_in[6];
  const float* Wfh   = (const float*)d_in[7];
  const float* bfh   = (const float*)d_in[8];
  const float* Wattn = (const float*)d_in[9];
  const float* battn = (const float*)d_in[10];
  const float* Whw   = (const float*)d_in[11];
  const float* bwh   = (const float*)d_in[12];
  const float* Wp    = (const float*)d_in[13];
  const float* bwp   = (const float*)d_in[14];
  const int*   lin   = (const int*)d_in[15];
  const int*   rin   = (const int*)d_in[16];
  float* out = (float*)d_out;

  char* p = (char*)d_ws;
  __bf16* WxP   = (__bf16*)carve(p, (size_t)HID * GATES * 2);       // 8 MB
  __bf16* WhT   = (__bf16*)carve(p, (size_t)HID * GATES * 2);       // 8 MB
  float*  biasX = (float*) carve(p, (size_t)GATES * 4);
  __bf16* X2    = (__bf16*)carve(p, (size_t)2 * SEQ * HID * 2);     // 2 MB
  float*  PX2   = (float*) carve(p, (size_t)2 * SEQ * GATES * 4);   // 16 MB
  float*  hbuf  = (float*) carve(p, (size_t)2 * 4 * HID * 4);
  float*  hFL   = (float*) carve(p, (size_t)SEQ * HID * 4);
  float*  hBL   = (float*) carve(p, (size_t)SEQ * HID * 4);
  float*  hFR   = (float*) carve(p, (size_t)SEQ * HID * 4);
  float*  hBR   = (float*) carve(p, (size_t)SEQ * HID * 4);
  float*  hl    = (float*) carve(p, (size_t)SEQ * HID * 4);
  float*  hr    = (float*) carve(p, (size_t)SEQ * HID * 4);
  unsigned* ctr = (unsigned*)carve(p, 256);

  const size_t nW = (size_t)HID * GATES;            // 4,194,304
  gather_init_kernel<<<(2 * SEQ * HID + 255) / 256, 256, 0, stream>>>(
      emb, lin, rin, X2, hbuf, ctr);
  pack_wx_kernel<<<(int)((nW + 255) / 256), 256, 0, stream>>>(Wioux, Wfx, WxP);
  conv_wh_kernel<<<(int)((nW + 255) / 256), 256, 0, stream>>>(
      Wiouh, Wfh, bioux, biouh, bfx, bfh, WhT, biasX);

  // 16384 C tiles, 8 waves (one tile each) per 256-thread block
  xproj_gemm_kernel<<<2048, 256, 0, stream>>>(X2, WxP, biasX, PX2);

  lstm_scan_kernel<<<NWG, 256, 0, stream>>>(
      WhT, PX2, PX2 + (size_t)SEQ * GATES, hbuf, hFL, hBL, hFR, hBR, ctr);

  combine_kernel<<<(SEQ * HID + 255) / 256, 256, 0, stream>>>(hFL, hBL, hl);
  combine_kernel<<<(SEQ * HID + 255) / 256, 256, 0, stream>>>(hFR, hBR, hr);

  tail_kernel<<<1, 1024, 0, stream>>>(hl, hr, Wattn, battn, Whw, bwh, Wp, bwp, out);
}
